// DyResConv_Inf_67817533604504
// MI455X (gfx1250) — matrix operand order repaired and checked by reading, not compiled, not run on hardware
//
#include <hip/hip_runtime.h>
#include <hip/hip_bf16.h>

typedef __attribute__((ext_vector_type(16))) _Float16 v16h;
typedef __attribute__((ext_vector_type(8)))  _Float16 v8h;
typedef __attribute__((ext_vector_type(8)))  float    v8f;

#define C_IN   768
#define HW     14400
#define HIMG   120
#define KTOT   6912      // 9 * 768, k = (kh*3+kw)*768 + i
#define EXP    3
#define SQ     48

// ---------------------------------------------------------------------------
// Kernel 1: adaptive pools (1x1, 3x3, 5x5 exact block means) + bicubic 3->5.
// ---------------------------------------------------------------------------
__global__ __launch_bounds__(256) void pool_kernel(const float* __restrict__ x,
                                                   float* __restrict__ att) {
    const int c = blockIdx.x;
    const int t = threadIdx.x;
    __shared__ float s3[9];
    __shared__ float s5[25];
    if (t < 9)  s3[t] = 0.f;
    if (t < 25) s5[t] = 0.f;
    __syncthreads();
    const float* xc = x + (size_t)c * HW;
    for (int p = t; p < HW; p += 256) {
        float v = xc[p];
        int h = p / HIMG, w = p % HIMG;
        atomicAdd(&s3[(h / 40) * 3 + (w / 40)], v);   // ds_add_f32
        atomicAdd(&s5[(h / 24) * 5 + (w / 24)], v);
    }
    __syncthreads();
    if (t < 25) {
        // PyTorch bicubic (a=-0.75), align_corners=False, border replicate, 3->5
        const float M35[5][3] = {
            { 1.096f, -0.096f,  0.000f},
            { 0.612f,  0.460f, -0.072f},
            { 0.000f,  1.000f,  0.000f},
            {-0.072f,  0.460f,  0.612f},
            { 0.000f, -0.096f,  1.096f}};
        float tot = 0.f;
        #pragma unroll
        for (int j = 0; j < 9; ++j) tot += s3[j];
        const float a1  = tot * (1.f / 14400.f);
        const float a5v = s5[t] * (1.f / 576.f);
        const int pr = t / 5, qc = t % 5;
        float a3u = 0.f;
        #pragma unroll
        for (int i = 0; i < 3; ++i)
            #pragma unroll
            for (int j = 0; j < 3; ++j)
                a3u += M35[pr][i] * M35[qc][j] * (s3[i * 3 + j] * (1.f / 1600.f));
        att[(size_t)c * 25 + t]              = a1;
        att[(size_t)(C_IN + c) * 25 + t]     = a3u;
        att[(size_t)(2 * C_IN + c) * 25 + t] = a5v;
    }
}

// ---------------------------------------------------------------------------
// Kernel 2: routing MLP -> 2304 sigmoid gates.
// ---------------------------------------------------------------------------
__global__ __launch_bounds__(256) void route_mlp(const float* __restrict__ att,
                                                 const float* __restrict__ w_pw1,
                                                 const float* __restrict__ w_dw1,
                                                 const float* __restrict__ w_dw2,
                                                 const float* __restrict__ w_pw2,
                                                 float* __restrict__ gates) {
    __shared__ float h1[SQ][25];
    __shared__ float h2[SQ][9];
    __shared__ float h3[SQ];
    const int t = threadIdx.x;
    for (int idx = t; idx < SQ * 25; idx += 256) {
        int s = idx / 25, p = idx % 25;
        float acc = 0.f;
        for (int c = 0; c < 3 * C_IN; ++c)
            acc += w_pw1[(size_t)s * (3 * C_IN) + c] * att[(size_t)c * 25 + p];
        h1[s][p] = fmaxf(acc, 0.f);
    }
    __syncthreads();
    for (int idx = t; idx < SQ * 9; idx += 256) {
        int s = idx / 9, q = idx % 9, ph = q / 3, pw = q % 3;
        float acc = 0.f;
        #pragma unroll
        for (int dh = 0; dh < 3; ++dh)
            #pragma unroll
            for (int dw = 0; dw < 3; ++dw)
                acc += h1[s][(ph + dh) * 5 + (pw + dw)] * w_dw1[s * 9 + dh * 3 + dw];
        h2[s][q] = fmaxf(acc, 0.f);
    }
    __syncthreads();
    for (int s = t; s < SQ; s += 256) {
        float acc = 0.f;
        #pragma unroll
        for (int q = 0; q < 9; ++q) acc += h2[s][q] * w_dw2[s * 9 + q];
        h3[s] = fmaxf(acc, 0.f);
    }
    __syncthreads();
    for (int o = t; o < EXP * C_IN; o += 256) {
        float acc = 0.f;
        #pragma unroll
        for (int s = 0; s < SQ; ++s) acc += w_pw2[(size_t)o * SQ + s] * h3[s];
        gates[o] = 1.f / (1.f + __expf(-acc));
    }
}

// ---------------------------------------------------------------------------
// Kernel 3: fused-weight synthesis -> f16, layout wA[o][k], k = r*768 + i.
// ---------------------------------------------------------------------------
__global__ __launch_bounds__(256) void synth_kernel(const float* __restrict__ convs,
                                                    const float* __restrict__ gates,
                                                    _Float16* __restrict__ wA) {
    const size_t idx = (size_t)blockIdx.x * 256 + threadIdx.x;
    if (idx >= (size_t)C_IN * KTOT) return;
    const int o = (int)(idx / KTOT);
    const int k = (int)(idx % KTOT);
    const int r = k / C_IN;      // kh*3 + kw
    const int i = k % C_IN;
    float acc = 0.f;
    #pragma unroll
    for (int e = 0; e < EXP; ++e)
        acc += gates[e * C_IN + o] *
               convs[(((size_t)e * C_IN + o) * C_IN + i) * 9 + r];
    wA[idx] = (_Float16)acc;
}

// ---------------------------------------------------------------------------
// Kernel 4: x f32 -> f16 (stays L2-resident for the im2col re-reads).
// ---------------------------------------------------------------------------
__global__ __launch_bounds__(256) void cvtx_kernel(const float* __restrict__ x,
                                                   _Float16* __restrict__ xh) {
    const size_t idx = (size_t)blockIdx.x * 256 + threadIdx.x;
    if (idx < (size_t)C_IN * HW) xh[idx] = (_Float16)x[idx];
}

// ---------------------------------------------------------------------------
// Kernel 5: implicit-GEMM 3x3 conv via v_wmma_f32_16x16x32_f16.
// M=768, N=14400, K=6912. Block tile 128x256, K-tile 32, LDS double-buffered.
// 8 waves (2x4); each wave owns 64x64 = 4x4 fragments -> 16 WMMA / K-step,
// matched 1:1 with 16 ds_load_b128 fragment reads. Global->register prefetch
// of tile k+1 is issued before the WMMA block of tile k; one barrier/iter.
// Since 768 % 32 == 0, each K-tile lies inside one filter tap r -> uniform
// (kh,kw) shift and bounds check for the whole tile.
// ---------------------------------------------------------------------------
#define LDK 40          // padded LDS row stride (halves); 80B keeps 16B align
#define NT  256
#define NKSTEP (KTOT / 32)   // 216

__global__ __launch_bounds__(256) void gemm_kernel(const _Float16* __restrict__ wA,
                                                   const _Float16* __restrict__ xh,
                                                   float* __restrict__ out) {
    __shared__ __align__(16) _Float16 lA[2][128][LDK];   // [buf][o][k]
    __shared__ __align__(16) _Float16 lB[2][NT][LDK];    // [buf][n][k]
    const int tid   = threadIdx.x;
    const int oBase = blockIdx.x * 128;
    const int pBase = blockIdx.y * NT;
    const int wave  = tid >> 5, lane = tid & 31;
    const int wm    = wave >> 2, wn = wave & 3;          // 2 x 4 wave grid
    const int lm    = lane & 15, half = lane >> 4;

    // A staging: rep rp covers row (tid>>2)+rp*64, 8 halves at ako
    const int arow0 = tid >> 2;
    const int ako   = (tid & 3) << 3;
    // B staging: rep rp covers n = tid, k-section rp*16 (all lanes of a wave
    // read consecutive pixels of the SAME channel -> coalesced 64B/instr)
    const int bn  = tid;
    const int bp  = pBase + bn;
    const int bph = bp / HIMG, bpw = bp % HIMG;
    const bool bpv = bp < HW;

    v8h ra[2];        // A tile registers (2 x 16B)
    v8h rb[2][2];     // B tile registers (2 sections x 16 halves)

    auto stage_global = [&](int k0) {
        #pragma unroll
        for (int rp = 0; rp < 2; ++rp)
            ra[rp] = *(const v8h*)(wA + (size_t)(oBase + arow0 + rp * 64) * KTOT + k0 + ako);
        const int r  = k0 / C_IN;               // filter tap, uniform over tile
        const int ib = k0 % C_IN;               // first input channel
        const int hh = bph + (r / 3) - 1;
        const int ww = bpw + (r % 3) - 1;
        if (bpv && (unsigned)hh < (unsigned)HIMG && (unsigned)ww < (unsigned)HIMG) {
            #pragma unroll
            for (int rp = 0; rp < 2; ++rp) {
                const _Float16* src = xh + (size_t)(ib + rp * 16) * HW + hh * HIMG + ww;
                #pragma unroll
                for (int e = 0; e < 8; ++e) {
                    rb[rp][0][e] = src[(size_t)e * HW];
                    rb[rp][1][e] = src[(size_t)(e + 8) * HW];
                }
            }
        } else {
            #pragma unroll
            for (int rp = 0; rp < 2; ++rp)
                #pragma unroll
                for (int e = 0; e < 8; ++e) {
                    rb[rp][0][e] = (_Float16)0.f;
                    rb[rp][1][e] = (_Float16)0.f;
                }
        }
    };
    auto stage_lds = [&](int buf) {
        #pragma unroll
        for (int rp = 0; rp < 2; ++rp) {
            *(v8h*)&lA[buf][arow0 + rp * 64][ako] = ra[rp];
            *(v8h*)&lB[buf][bn][rp * 16]          = rb[rp][0];
            *(v8h*)&lB[buf][bn][rp * 16 + 8]      = rb[rp][1];
        }
    };

    v8f acc[4][4];
    const v8f zero = {};
    #pragma unroll
    for (int fm = 0; fm < 4; ++fm)
        #pragma unroll
        for (int fn = 0; fn < 4; ++fn) acc[fm][fn] = zero;

    stage_global(0);
    stage_lds(0);
    __syncthreads();

    for (int kt = 0; kt < NKSTEP; ++kt) {
        const int cur  = kt & 1;
        const bool more = (kt + 1) < NKSTEP;
        if (more) {
            stage_global((kt + 1) * 32);        // prefetch next tile to regs
            if (kt + 2 < NKSTEP)                // and hint the one after that
                __builtin_prefetch(wA + (size_t)(oBase + arow0) * KTOT + (kt + 2) * 32, 0, 3);
        }

        // B fragments (contiguous k order, same as A: elem e <-> K = half*16 + e, col = lane%16)
        v16h bf[4];
        #pragma unroll
        for (int fn = 0; fn < 4; ++fn) {
            int col = wn * 64 + fn * 16 + lm;
            v8h b0 = *(const v8h*)&lB[cur][col][half * 16];
            v8h b1 = *(const v8h*)&lB[cur][col][half * 16 + 8];
            bf[fn] = __builtin_shufflevector(b0, b1, 0,1,2,3,4,5,6,7,8,9,10,11,12,13,14,15);
        }
        // A fragment per fm, reused across 4 consecutive WMMAs
        #pragma unroll
        for (int fm = 0; fm < 4; ++fm) {
            int row = wm * 64 + fm * 16 + lm;
            v8h a0 = *(const v8h*)&lA[cur][row][half * 16];      // K = half*16+0..7
            v8h a1 = *(const v8h*)&lA[cur][row][half * 16 + 8];  // K = half*16+8..15
            v16h af = __builtin_shufflevector(a0, a1, 0,1,2,3,4,5,6,7,8,9,10,11,12,13,14,15);
            #pragma unroll
            for (int fn = 0; fn < 4; ++fn)
                acc[fm][fn] = __builtin_amdgcn_wmma_f32_16x16x32_f16(
                    false, af, false, bf[fn], (short)0, acc[fm][fn], false, false);
        }

        if (more) {
            stage_lds(cur ^ 1);                 // fill ping-pong buffer
            __syncthreads();                    // single barrier per K-step
        }
    }

    // epilogue: C/D layout -> lane holds col N = lane%16, rows M = v + 8*half
    #pragma unroll
    for (int fm = 0; fm < 4; ++fm)
        #pragma unroll
        for (int fn = 0; fn < 4; ++fn) {
            int pcol = pBase + wn * 64 + fn * 16 + lm;
            if (pcol < HW) {
                int ob = oBase + wm * 64 + fm * 16 + half * 8;
                #pragma unroll
                for (int v = 0; v < 8; ++v)
                    out[(size_t)(ob + v) * HW + pcol] = acc[fm][fn][v];
            }
        }
}

// ---------------------------------------------------------------------------
// kernel_launch: inputs (setup_inputs order): x, convs, w_pw1, w_dw1, w_dw2, w_pw2
// ---------------------------------------------------------------------------
extern "C" void kernel_launch(void* const* d_in, const int* in_sizes, int n_in,
                              void* d_out, int out_size, void* d_ws, size_t ws_size,
                              hipStream_t stream) {
    const float* x      = (const float*)d_in[0];
    const float* convs  = (const float*)d_in[1];
    const float* w_pw1  = (const float*)d_in[2];
    const float* w_dw1  = (const float*)d_in[3];
    const float* w_dw2  = (const float*)d_in[4];
    const float* w_pw2  = (const float*)d_in[5];
    float* y = (float*)d_out;

    // workspace layout (256B-aligned offsets), total ~33 MB
    char* ws = (char*)d_ws;
    float*    att   = (float*)ws;                                  // 2304*25 f32 = 230400 B
    float*    gates = (float*)(ws + 230400);                       // 2304 f32   =   9216 B
    _Float16* wA    = (_Float16*)(ws + 239616);                    // 768*6912 f16 = 10616832 B
    _Float16* xh    = (_Float16*)(ws + 239616 + 10616832);         // 768*14400 f16 = 22118400 B

    pool_kernel<<<C_IN, 256, 0, stream>>>(x, att);
    route_mlp  <<<1, 256, 0, stream>>>(att, w_pw1, w_dw1, w_dw2, w_pw2, gates);
    synth_kernel<<<(int)(((size_t)C_IN * KTOT + 255) / 256), 256, 0, stream>>>(convs, gates, wA);
    cvtx_kernel <<<(int)(((size_t)C_IN * HW + 255) / 256), 256, 0, stream>>>(x, xh);
    gemm_kernel <<<dim3(6, (HW + NT - 1) / NT), 256, 0, stream>>>(wA, xh, y);
}
